// FlaxRwkvSelfAttention_549755814276
// MI455X (gfx1250) — matrix operand order repaired and checked by reading, not compiled, not run on hardware
//
#include <hip/hip_runtime.h>
#include <cstddef>

// Problem dims (fixed by reference)
#define BSZ 4
#define TSZ 2048
#define HSZ 2048
#define MSZ (BSZ * TSZ)      // 8192 rows for every GEMM
#define KDIM HSZ             // all four GEMMs are [8192 x 2048] @ [2048 x 2048]
#define NDIM HSZ

typedef __attribute__((ext_vector_type(16))) _Float16 v16h;
typedef __attribute__((ext_vector_type(8)))  _Float16 v8h;
typedef __attribute__((ext_vector_type(4)))  _Float16 v4h;
typedef __attribute__((ext_vector_type(8)))  float    v8f;
typedef __attribute__((ext_vector_type(4)))  int      v4i;

union Frag { v16h v; v8h h[2]; };

#define AS1 __attribute__((address_space(1)))
#define AS3 __attribute__((address_space(3)))

#if defined(__AMDGCN__) && __has_builtin(__builtin_amdgcn_global_load_async_to_lds_b128)
#define HAVE_ASYNC 1
#else
#define HAVE_ASYNC 0
#endif

__device__ __forceinline__ void wait_async0() {
#if defined(__AMDGCN__)
#if __has_builtin(__builtin_amdgcn_s_wait_asynccnt)
    __builtin_amdgcn_s_wait_asynccnt(0);
#else
    asm volatile("s_wait_asynccnt 0x0" ::: "memory");
#endif
#endif
}

// ---------------------------------------------------------------------------
// Weight prep: Wt[n][k] = (f16) W[k][n] — 32x32 LDS tile transpose, both
// global read and write coalesced.
// ---------------------------------------------------------------------------
__global__ __launch_bounds__(256) void transpose_f16_kernel(
    const float* __restrict__ W, _Float16* __restrict__ Wt) {
    __shared__ float tile[32][33];
    const int kb = blockIdx.x * 32;      // k tile base
    const int nb = blockIdx.y * 32;      // n tile base
    const int tx = threadIdx.x & 31;
    const int ty = threadIdx.x >> 5;     // 0..7
    #pragma unroll
    for (int j = ty; j < 32; j += 8)     // read rows k, consecutive n: coalesced
        tile[j][tx] = W[(size_t)(kb + j) * HSZ + nb + tx];
    __syncthreads();
    #pragma unroll
    for (int j = ty; j < 32; j += 8)     // write rows n, consecutive k: coalesced
        Wt[(size_t)(nb + j) * HSZ + kb + tx] = (_Float16)tile[tx][j];
}

// ---------------------------------------------------------------------------
// Time-mix: out[b,t,d] = h[b,t,d]*m[d] + h[b,t-1,d]*(1-m[d])  (f16 out)
// ---------------------------------------------------------------------------
__global__ __launch_bounds__(256) void mix_kernel(
    const float* __restrict__ hidden, const float* __restrict__ tm,
    _Float16* __restrict__ out) {
    size_t i4 = ((size_t)blockIdx.x * blockDim.x + threadIdx.x) * 4;
    int d = (int)(i4 & (HSZ - 1));
    int t = (int)((i4 >> 11) & (TSZ - 1));
    float4 h = *(const float4*)(hidden + i4);
    float4 s = make_float4(0.f, 0.f, 0.f, 0.f);
    if (t > 0) s = *(const float4*)(hidden + i4 - HSZ);
    float4 m = *(const float4*)(tm + d);
    v4h o;
    o[0] = (_Float16)(h.x * m.x + s.x * (1.0f - m.x));
    o[1] = (_Float16)(h.y * m.y + s.y * (1.0f - m.y));
    o[2] = (_Float16)(h.z * m.z + s.z * (1.0f - m.z));
    o[3] = (_Float16)(h.w * m.w + s.w * (1.0f - m.w));
    *(v4h*)(out + i4) = o;
}

// ---------------------------------------------------------------------------
// WMMA GEMM: D[M,N] (f32) = A[M,K] (f16, row-major) x Bt[N,K] (f16, N-major)
// 256 thr = 8 waves (4 M x 2 N). Block tile 128x128, wave tile 32x64 =
// 2x4 v_wmma_f32_16x16x32_f16. Double-buffered LDS fed by
// GLOBAL_LOAD_ASYNC_TO_LDS_B128. K-loop kept ROLLED (runtime buffer index)
// so accumulators keep one register assignment -> no v_mov/v_nop churn.
// ---------------------------------------------------------------------------
#define BM 128
#define BN 128
#define BK 32
#define KT (KDIM / BK)       // 64
#define LDT (BK + 8)         // 40 halves = 80B row stride: conflict-free, 16B aligned
#define ATILE (BM * LDT)     // halves per A buffer
#define BTILE (BN * LDT)     // halves per B buffer

__global__ __launch_bounds__(256) void gemm_f16_wmma(
    const _Float16* __restrict__ A, const _Float16* __restrict__ Bt,
    float* __restrict__ D)
{
    __shared__ __align__(16) _Float16 As[2 * ATILE];
    __shared__ __align__(16) _Float16 Bs[2 * BTILE];

    const int tid  = threadIdx.x;
    const int lane = tid & 31;
    const int w    = tid >> 5;          // 0..7
    const int wm   = w & 3;             // wave M slot (0..3), 32 rows each
    const int wn   = w >> 2;            // wave N slot (0..1), 64 cols each
    const bool hiHalf = (lane >= 16);
    const int mrow = lane & 15;
    const int m0 = blockIdx.y * BM;
    const int n0 = blockIdx.x * BN;

    // Per-thread staging chunks: A and B tiles are each 128 rows x 4 x 16B
    const int rA0 = tid >> 2,          cA0 = (tid & 3) * 8;
    const int rA1 = (tid + 256) >> 2,  cA1 = cA0;           // (tid+256)&3 == tid&3
    const _Float16* gA0 = A  + (size_t)(m0 + rA0) * KDIM + cA0;
    const _Float16* gA1 = A  + (size_t)(m0 + rA1) * KDIM + cA1;
    const _Float16* gB0 = Bt + (size_t)(n0 + rA0) * KDIM + cA0;
    const _Float16* gB1 = Bt + (size_t)(n0 + rA1) * KDIM + cA1;
    const int lA0 = rA0 * LDT + cA0;
    const int lA1 = rA1 * LDT + cA1;

    auto stage = [&](int buf, int k0) {
        _Float16* aDst = As + buf * ATILE;
        _Float16* bDst = Bs + buf * BTILE;
#if HAVE_ASYNC
        __builtin_amdgcn_global_load_async_to_lds_b128(
            (AS1 v4i*)(gA0 + k0), (AS3 v4i*)(aDst + lA0), 0, 0);
        __builtin_amdgcn_global_load_async_to_lds_b128(
            (AS1 v4i*)(gA1 + k0), (AS3 v4i*)(aDst + lA1), 0, 0);
        __builtin_amdgcn_global_load_async_to_lds_b128(
            (AS1 v4i*)(gB0 + k0), (AS3 v4i*)(bDst + lA0), 0, 0);
        __builtin_amdgcn_global_load_async_to_lds_b128(
            (AS1 v4i*)(gB1 + k0), (AS3 v4i*)(bDst + lA1), 0, 0);
#else
        *(v8h*)(aDst + lA0) = *(const v8h*)(gA0 + k0);
        *(v8h*)(aDst + lA1) = *(const v8h*)(gA1 + k0);
        *(v8h*)(bDst + lA0) = *(const v8h*)(gB0 + k0);
        *(v8h*)(bDst + lA1) = *(const v8h*)(gB1 + k0);
#endif
    };

    v8f acc[2][4] = {};

    stage(0, 0);
#if HAVE_ASYNC
    wait_async0();
#endif
    __syncthreads();

    int cur = 0;
    #pragma clang loop unroll(disable)
    for (int kt = 0; kt < KT; ++kt) {
        if (kt + 1 < KT) stage(cur ^ 1, (kt + 1) * BK);

        const _Float16* aSrc = As + cur * ATILE;
        const _Float16* bSrc = Bs + cur * BTILE;

        // A fragments (ISA 16-bit A layout: lanes 0-15 K0-7/K16-23,
        // lanes 16-31 K8-15/K24-31)
        Frag a0, a1;
        {
            int r0 = (wm * 32 + mrow) * LDT + (hiHalf ? 8 : 0);
            a0.h[0] = *(const v8h*)(aSrc + r0);
            a0.h[1] = *(const v8h*)(aSrc + r0 + 16);
            a1.h[0] = *(const v8h*)(aSrc + r0 + 16 * LDT);
            a1.h[1] = *(const v8h*)(aSrc + r0 + 16 * LDT + 16);
        }
        // One live B fragment at a time: load b, fire both MMAs that use it.
        #pragma unroll
        for (int sn = 0; sn < 4; ++sn) {
            Frag b;
            int rb = (wn * 64 + sn * 16 + mrow) * LDT + (hiHalf ? 8 : 0);
            b.h[0] = *(const v8h*)(bSrc + rb);
            b.h[1] = *(const v8h*)(bSrc + rb + 16);
            acc[0][sn] = __builtin_amdgcn_wmma_f32_16x16x32_f16(
                false, a0.v, false, b.v, (short)0, acc[0][sn], false, false);
            acc[1][sn] = __builtin_amdgcn_wmma_f32_16x16x32_f16(
                false, a1.v, false, b.v, (short)0, acc[1][sn], false, false);
        }

#if HAVE_ASYNC
        // Keep the asynccnt wait AFTER the MMA clump (WMMAs are not memory
        // ops, so the side-effecting wait would otherwise be scheduled above
        // them and serialize compute with the next tile's loads).
        {
            float f0 = acc[0][0][0], f1 = acc[1][3][7];
            asm volatile("" :: "v"(f0), "v"(f1));
        }
        wait_async0();
#endif
        __syncthreads();
        cur ^= 1;
    }

    // Epilogue: C/D layout -> VGPR r holds M = r + (lane>=16 ? 8 : 0), N = lane%16
    #pragma unroll
    for (int sm = 0; sm < 2; ++sm)
        #pragma unroll
        for (int sn = 0; sn < 4; ++sn) {
            int gn = n0 + wn * 64 + sn * 16 + mrow;
            int gmBase = m0 + wm * 32 + sm * 16 + (hiHalf ? 8 : 0);
            #pragma unroll
            for (int r8 = 0; r8 < 8; ++r8)
                D[(size_t)(gmBase + r8) * NDIM + gn] = acc[sm][sn][r8];
        }
}

// ---------------------------------------------------------------------------
// WKV as a chunked scan. NUM' = e^td NUM + e^k v ; DEN' = e^td DEN + e^k is
// affine in state, so: pass1 computes per-chunk partial sums (zero init) in
// (value, max) stabilized form; pass2 serially combines 16 chunk states;
// pass3 replays each chunk from its exact incoming state, emitting outputs.
// Serial depth: 2048 -> 128 + 16 + 128. Parallelism: 8192 -> 131072 threads.
// ---------------------------------------------------------------------------
#define NCHUNK 16
#define CLEN (TSZ / NCHUNK)              // 128
#define SPLANE (NCHUNK * BSZ * HSZ)      // 131072 states per plane

__global__ __launch_bounds__(256) void wkv_pass1(
    const float* __restrict__ kbuf, const float* __restrict__ vbuf,
    const float* __restrict__ time_decay, float* __restrict__ st)
{
    int idx = blockIdx.x * blockDim.x + threadIdx.x;   // ((c*B)+b)*H + d
    int d  = idx & (HSZ - 1);
    int bc = idx >> 11;
    int b  = bc & (BSZ - 1);
    int c  = bc >> 2;
    float td = -__expf(time_decay[d]);

    float p = 0.0f, q = 0.0f, mc = -1e38f;
    size_t base = (size_t)b * TSZ * HSZ + (size_t)c * CLEN * HSZ + d;
    for (int s = 0; s < CLEN; ++s) {
        size_t off = base + (size_t)s * HSZ;
        float ks = kbuf[off], vs = vbuf[off];
        float mfs = fmaxf(mc + td, ks);
        float e1  = __expf(mc + td - mfs);
        float e2  = __expf(ks - mfs);
        p = e1 * p + e2 * vs;
        q = e1 * q + e2;
        mc = mfs;
    }
    st[idx]              = p;
    st[idx + SPLANE]     = q;
    st[idx + 2 * SPLANE] = mc;
}

__global__ __launch_bounds__(256) void wkv_pass2(
    const float* __restrict__ time_decay, const float* __restrict__ st,
    float* __restrict__ init)
{
    int idx = blockIdx.x * blockDim.x + threadIdx.x;   // 0..B*H-1
    int d = idx & (HSZ - 1);
    int b = idx >> 11;
    float td  = -__expf(time_decay[d]);
    float dec = td * (float)CLEN;                      // log-decay per chunk

    float num = 0.0f, den = 0.0f, mx = -1e38f;
    for (int c = 0; c < NCHUNK; ++c) {
        int sidx = ((c * BSZ) + b) * HSZ + d;
        init[sidx]              = num;
        init[sidx + SPLANE]     = den;
        init[sidx + 2 * SPLANE] = mx;
        float p  = st[sidx];
        float q  = st[sidx + SPLANE];
        float mc = st[sidx + 2 * SPLANE];
        float mxs = mx + dec;
        float m  = fmaxf(mxs, mc);
        float ea = __expf(mxs - m);
        float eb = __expf(mc - m);
        num = ea * num + eb * p;
        den = ea * den + eb * q;
        mx  = m;
    }
}

__global__ __launch_bounds__(256) void wkv_pass3(
    const float* __restrict__ kbuf, const float* __restrict__ vbuf,
    const float* __restrict__ rbuf, const float* __restrict__ time_decay,
    const float* __restrict__ time_first, const float* __restrict__ init,
    _Float16* __restrict__ y)
{
    int idx = blockIdx.x * blockDim.x + threadIdx.x;   // ((c*B)+b)*H + d
    int d  = idx & (HSZ - 1);
    int bc = idx >> 11;
    int b  = bc & (BSZ - 1);
    int c  = bc >> 2;
    float td = -__expf(time_decay[d]);
    float tf = time_first[d];

    float num = init[idx];
    float den = init[idx + SPLANE];
    float mx  = init[idx + 2 * SPLANE];

    size_t base = (size_t)b * TSZ * HSZ + (size_t)c * CLEN * HSZ + d;
    for (int s = 0; s < CLEN; ++s) {
        size_t off = base + (size_t)s * HSZ;
        float kt = kbuf[off];
        float vt = vbuf[off];
        float mfo = fmaxf(mx, kt + tf);
        float e1  = __expf(mx - mfo);
        float e2  = __expf(kt + tf - mfo);
        float out = (e1 * num + e2 * vt) / (e1 * den + e2);
        float mfs = fmaxf(mx + td, kt);
        float e1s = __expf(mx + td - mfs);
        float e2s = __expf(kt - mfs);
        num = e1s * num + e2s * vt;
        den = e1s * den + e2s;
        mx  = mfs;
        float rr  = rbuf[off];
        float sig = 1.0f / (1.0f + __expf(-rr));
        y[off] = (_Float16)(sig * out);
    }
}

// ---------------------------------------------------------------------------
extern "C" void kernel_launch(void* const* d_in, const int* in_sizes, int n_in,
                              void* d_out, int out_size, void* d_ws, size_t ws_size,
                              hipStream_t stream) {
    const float* hidden = (const float*)d_in[0];
    const float* tdec   = (const float*)d_in[1];
    const float* tfir   = (const float*)d_in[2];
    const float* tmk    = (const float*)d_in[3];
    const float* tmv    = (const float*)d_in[4];
    const float* tmr    = (const float*)d_in[5];
    const float* Wk     = (const float*)d_in[6];
    const float* Wv     = (const float*)d_in[7];
    const float* Wr     = (const float*)d_in[8];
    const float* Wo     = (const float*)d_in[9];

    const size_t HH = (size_t)HSZ * HSZ;        // 4,194,304
    const size_t MK = (size_t)MSZ * HSZ;        // 16,777,216

    // Workspace layout (~259 MiB):
    //   [0)       4 x HH f16 : transposed weights
    //   [32MiB)   MK f16     : mixed input / fused (r*wkv) buffer (reused)
    //   [64MiB)   MK f32     : k
    //   [128MiB)  MK f32     : v
    //   [192MiB)  MK f32     : r (pre-sigmoid)
    //   [256MiB)  3*SPLANE f32 : chunk partial states
    //   [+1.5MiB) 3*SPLANE f32 : chunk initial states
    char* ws = (char*)d_ws;
    _Float16* Wtk = (_Float16*)ws;
    _Float16* Wtv = Wtk + HH;
    _Float16* Wtr = Wtv + HH;
    _Float16* Wto = Wtr + HH;
    _Float16* mixed = (_Float16*)(ws + 4 * HH * sizeof(_Float16));
    float* kbuf = (float*)(ws + 4 * HH * sizeof(_Float16) + MK * sizeof(_Float16));
    float* vbuf = kbuf + MK;
    float* rbuf = vbuf + MK;
    float* stbuf   = rbuf + MK;
    float* initbuf = stbuf + 3 * (size_t)SPLANE;
    float* outp = (float*)d_out;

    dim3 tgrid(HSZ / 32, HSZ / 32);             // 64 x 64
    const int mixGrid = (int)(MK / 4 / 256);    // 16384
    dim3 ggrid(NDIM / BN, MSZ / BM);            // 16 x 64 blocks

    // 1) weight prep (f16, N-major)
    transpose_f16_kernel<<<tgrid, 256, 0, stream>>>(Wk, Wtk);
    transpose_f16_kernel<<<tgrid, 256, 0, stream>>>(Wv, Wtv);
    transpose_f16_kernel<<<tgrid, 256, 0, stream>>>(Wr, Wtr);
    transpose_f16_kernel<<<tgrid, 256, 0, stream>>>(Wo, Wto);

    // 2) k = mix_k(hidden) @ Wk
    mix_kernel<<<mixGrid, 256, 0, stream>>>(hidden, tmk, mixed);
    gemm_f16_wmma<<<ggrid, 256, 0, stream>>>(mixed, Wtk, kbuf);
    // 3) v = mix_v(hidden) @ Wv
    mix_kernel<<<mixGrid, 256, 0, stream>>>(hidden, tmv, mixed);
    gemm_f16_wmma<<<ggrid, 256, 0, stream>>>(mixed, Wtv, vbuf);
    // 4) r_raw = mix_r(hidden) @ Wr   (sigmoid fused into wkv pass3)
    mix_kernel<<<mixGrid, 256, 0, stream>>>(hidden, tmr, mixed);
    gemm_f16_wmma<<<ggrid, 256, 0, stream>>>(mixed, Wtr, rbuf);

    // 5) chunked WKV scan + receptance gate -> f16 (reuses `mixed`)
    wkv_pass1<<<SPLANE / 256, 256, 0, stream>>>(kbuf, vbuf, tdec, stbuf);
    wkv_pass2<<<(BSZ * HSZ) / 256, 256, 0, stream>>>(tdec, stbuf, initbuf);
    wkv_pass3<<<SPLANE / 256, 256, 0, stream>>>(kbuf, vbuf, rbuf, tdec, tfir,
                                                initbuf, mixed);
    // 6) out = y @ Wo
    gemm_f16_wmma<<<ggrid, 256, 0, stream>>>(mixed, Wto, outp);
}
